// SemanticEncoder_41583873360678
// MI455X (gfx1250) — compile-verified, matrix operand order checked by reading, not run on hardware
//
#include <hip/hip_runtime.h>
#include <hip/hip_bf16.h>

// ---------------------------------------------------------------------------
// Residual Vector Quantizer, fused, CDNA5 (gfx1250, wave32, WMMA bf16).
//
// Key algebra: argmin_c d2 == argmin_c ( -2 * residual . G_l[c] + cbn_l[c] )
//   where G_l = (cb_l @ W_l^T) @ W_l   [32,1024]   (precomputed, bf16)
//         cbn_l[c] = || (cb_l @ W_l^T)[c] ||^2     (precomputed, fp32)
// So each level is ONE [B,1024]x[1024,32] GEMM -> argmin -> gather/update.
// Residual lives in LDS as bf16 (only influences index selection); the final
// output q = sum cb_l[idx_l] is re-gathered in fp32, so output precision is
// unaffected by the bf16 residual.
// ---------------------------------------------------------------------------

typedef __attribute__((ext_vector_type(16))) __bf16 v16bf;
typedef __attribute__((ext_vector_type(8)))  __bf16 v8bf;
typedef __attribute__((ext_vector_type(4)))  __bf16 v4bf;
typedef __attribute__((ext_vector_type(8)))  float  v8f;
typedef __attribute__((ext_vector_type(4)))  float  v4f;

#define NLVL   4
#define CBSZ   32
#define DIM    1024
#define NBATCH 65536

union ABFrag { v16bf v; v8bf h[2]; };

// ---------------- prep: cbproj[l][c1][c2] = cb[l][c1] . W[l][c2] -----------
__global__ void rvq_prep_cbproj(const float* __restrict__ Ws,
                                const float* __restrict__ cbs,
                                float* __restrict__ cbproj) {
    const int warp = (blockIdx.x * blockDim.x + threadIdx.x) >> 5;
    const int lane = threadIdx.x & 31;
    if (warp >= NLVL * CBSZ * CBSZ) return;
    const int l  = warp >> 10;
    const int c1 = (warp >> 5) & 31;
    const int c2 = warp & 31;
    const float* a = cbs + ((size_t)l * CBSZ + c1) * DIM;
    const float* b = Ws  + ((size_t)l * CBSZ + c2) * DIM;
    float s = 0.f;
    for (int d = lane; d < DIM; d += 32) s += a[d] * b[d];
    for (int off = 16; off > 0; off >>= 1) s += __shfl_xor(s, off, 32);
    if (lane == 0) cbproj[warp] = s;
}

// ---------------- prep: G[l][c][d] = sum_c2 cbproj[l][c][c2] * W[l][c2][d] -
__global__ void rvq_prep_G(const float* __restrict__ Ws,
                           const float* __restrict__ cbproj,
                           __bf16* __restrict__ G) {
    const int t = blockIdx.x * blockDim.x + threadIdx.x;
    if (t >= NLVL * CBSZ * DIM) return;
    const int d = t & (DIM - 1);
    const int c = (t >> 10) & 31;
    const int l = t >> 15;
    const float* cp = cbproj + (l * CBSZ + c) * CBSZ;
    const float* w  = Ws + (size_t)l * CBSZ * DIM + d;
    float s = 0.f;
#pragma unroll
    for (int c2 = 0; c2 < CBSZ; ++c2) s += cp[c2] * w[(size_t)c2 * DIM];
    G[t] = (__bf16)s;
}

// ---------------- prep: cbn[l][c] = ||cbproj[l][c]||^2 ---------------------
__global__ void rvq_prep_cbn(const float* __restrict__ cbproj,
                             float* __restrict__ cbn) {
    const int t = threadIdx.x;
    if (t < NLVL * CBSZ) {
        const float* cp = cbproj + t * CBSZ;
        float s = 0.f;
#pragma unroll
        for (int c2 = 0; c2 < CBSZ; ++c2) s += cp[c2] * cp[c2];
        cbn[t] = s;
    }
}

// ---------------- zero the 5 loss accumulators (atomics target) ------------
__global__ void rvq_zero(float* __restrict__ loss) {
    if (threadIdx.x < 5) loss[threadIdx.x] = 0.f;
}

// ---------------- main fused kernel ----------------------------------------
// block = 128 threads (4 waves); each wave owns 16 batch rows end-to-end.
// LDS: residual bf16 [4][16][1024] + scores f32 [4][16][32] + idx [4][4][16]
__global__ __launch_bounds__(128)
void rvq_main(const float* __restrict__ x,
              const float* __restrict__ cbs,   // [4][32][1024] fp32
              const __bf16* __restrict__ G,    // [4][32][1024] bf16
              const float* __restrict__ cbn,   // [4][32] fp32
              float* __restrict__ out) {
    extern __shared__ char smem[];
    __bf16* resAll = (__bf16*)smem;                                   // 128 KB
    float*  scAll  = (float*)(smem + 4 * 16 * DIM * sizeof(__bf16));  // 8 KB
    int*    idxAll = (int*)((char*)scAll + 4 * 16 * 32 * sizeof(float));

    const int lane = threadIdx.x & 31;
    const int wv   = threadIdx.x >> 5;
    const int half = lane >> 4;      // wave32 lane half (WMMA fragment split)
    const int lr   = lane & 15;

    __bf16* res = resAll + wv * 16 * DIM;
    float*  sc  = scAll  + wv * 16 * 32;
    int*    idx = idxAll + wv * NLVL * 16;

    const int rowBase = blockIdx.x * 64 + wv * 16;

    // ---- load x tile -> bf16 residual in LDS ----
    for (int m = 0; m < 16; ++m) {
        const v4f* xr = (const v4f*)(x + (size_t)(rowBase + m) * DIM);
        v4bf* rr = (v4bf*)(res + m * DIM);
        for (int d4 = lane; d4 < DIM / 4; d4 += 32) {
            v4f v = xr[d4];
            v4bf b;
            b[0] = (__bf16)v[0]; b[1] = (__bf16)v[1];
            b[2] = (__bf16)v[2]; b[3] = (__bf16)v[3];
            rr[d4] = b;
        }
    }
    __syncthreads();

    for (int l = 0; l < NLVL; ++l) {
        const __bf16* Gl  = G + (size_t)l * CBSZ * DIM;
        const __bf16* bn0 = Gl + (size_t)lr * DIM;        // B column n = lr
        const __bf16* bn1 = Gl + (size_t)(16 + lr) * DIM; // B column n = 16+lr
        const __bf16* arow = res + lr * DIM;              // A row m = lr
        __builtin_prefetch(bn0, 0, 3);
        __builtin_prefetch(bn1, 0, 3);

        // ---- S[m][c] = sum_k res[m][k] * Gl[c][k]  via 16x16x32 bf16 WMMA
        // A 16-bit 16x32 fragment: lanes 0-15 row m=lr hold K=kb..kb+7 (elems
        // 0-7) and K=kb+16..kb+23 (elems 8-15); lanes 16-31 hold the +8 / +24
        // K ranges (kb offset by half*8). B mirrors this over columns.
        v8f c0 = {};
        v8f c1 = {};
        for (int kc = 0; kc < DIM / 32; ++kc) {
            const int kb = kc * 32 + half * 8;
            ABFrag a, b0, b1;
            a.h[0]  = *(const v8bf*)(arow + kb);
            a.h[1]  = *(const v8bf*)(arow + kb + 16);
            b0.h[0] = *(const v8bf*)(bn0 + kb);
            b0.h[1] = *(const v8bf*)(bn0 + kb + 16);
            b1.h[0] = *(const v8bf*)(bn1 + kb);
            b1.h[1] = *(const v8bf*)(bn1 + kb + 16);
            c0 = __builtin_amdgcn_wmma_f32_16x16x32_bf16(
                     false, a.v, false, b0.v, (short)0, c0, false, false);
            c1 = __builtin_amdgcn_wmma_f32_16x16x32_bf16(
                     false, a.v, false, b1.v, (short)0, c1, false, false);
        }

        // ---- scatter accumulators to LDS: elem e -> row M = half*8 + e ----
        const int mofs = half * 8;
#pragma unroll
        for (int e = 0; e < 8; ++e) {
            sc[(mofs + e) * 32 + lr]      = c0[e];
            sc[(mofs + e) * 32 + 16 + lr] = c1[e];
        }
        __syncthreads();

        // ---- per-row argmin over 32 codes (first-min tie-break) ----
        if (lane < 16) {
            const float* cn   = cbn + l * CBSZ;
            const float* srow = sc + lane * 32;
            float best = cn[0] - 2.f * srow[0];
            int bi = 0;
#pragma unroll
            for (int c = 1; c < CBSZ; ++c) {
                float v = cn[c] - 2.f * srow[c];
                if (v < best) { best = v; bi = c; }
            }
            idx[l * 16 + lane] = bi;
        }
        __syncthreads();

        // ---- residual update: res[m] -= cb_l[idx[m]] (bf16, L2-hot cb) ----
        for (int m = 0; m < 16; ++m) {
            const int bi = idx[l * 16 + m];
            const v4f* cbr = (const v4f*)(cbs + ((size_t)l * CBSZ + bi) * DIM);
            v4bf* rr = (v4bf*)(res + m * DIM);
            for (int d4 = lane; d4 < DIM / 4; d4 += 32) {
                v4f  cv = cbr[d4];
                v4bf rv = rr[d4];
                v4bf nv;
                nv[0] = (__bf16)((float)rv[0] - cv[0]);
                nv[1] = (__bf16)((float)rv[1] - cv[1]);
                nv[2] = (__bf16)((float)rv[2] - cv[2]);
                nv[3] = (__bf16)((float)rv[3] - cv[3]);
                rr[d4] = nv;
            }
        }
        __syncthreads();
    }

    // ---- output pass: fp32 gathers, q_out, losses ----
    float recon_s = 0.f;
    float com_s[NLVL] = {0.f, 0.f, 0.f, 0.f};
    for (int m = 0; m < 16; ++m) {
        const size_t row = (size_t)rowBase + m;
        const v4f* xr   = (const v4f*)(x + row * DIM);
        v4f*       outr = (v4f*)(out + row * DIM);
        const v4f* r0 = (const v4f*)(cbs + ((size_t)0 * CBSZ + idx[0 * 16 + m]) * DIM);
        const v4f* r1 = (const v4f*)(cbs + ((size_t)1 * CBSZ + idx[1 * 16 + m]) * DIM);
        const v4f* r2 = (const v4f*)(cbs + ((size_t)2 * CBSZ + idx[2 * 16 + m]) * DIM);
        const v4f* r3 = (const v4f*)(cbs + ((size_t)3 * CBSZ + idx[3 * 16 + m]) * DIM);
        for (int d4 = lane; d4 < DIM / 4; d4 += 32) {
            v4f xv = xr[d4];
            v4f q0 = r0[d4], q1 = r1[d4], q2 = r2[d4], q3 = r3[d4];
            v4f q = q0 + q1 + q2 + q3;
            outr[d4] = q;
#pragma unroll
            for (int j = 0; j < 4; ++j) {
                float dx;
                dx = q[j]  - xv[j]; recon_s  += dx * dx;
                dx = q0[j] - xv[j]; com_s[0] += dx * dx;
                dx = q1[j] - xv[j]; com_s[1] += dx * dx;
                dx = q2[j] - xv[j]; com_s[2] += dx * dx;
                dx = q3[j] - xv[j]; com_s[3] += dx * dx;
            }
        }
    }

    // ---- indices (as float, per harness output flattening) ----
    float* idxOut = out + (size_t)NBATCH * DIM;
    if (lane < 16) {
#pragma unroll
        for (int l = 0; l < NLVL; ++l)
            idxOut[(size_t)l * NBATCH + rowBase + lane] = (float)idx[l * 16 + lane];
    }

    // ---- loss reduction: wave shfl-xor, then fp32 global atomics ----
    for (int off = 16; off > 0; off >>= 1) {
        recon_s += __shfl_xor(recon_s, off, 32);
#pragma unroll
        for (int l = 0; l < NLVL; ++l) com_s[l] += __shfl_xor(com_s[l], off, 32);
    }
    if (lane == 0) {
        float* loss = out + (size_t)NBATCH * DIM + (size_t)NLVL * NBATCH;
        const float inv = 1.0f / ((float)NBATCH * (float)DIM);
#pragma unroll
        for (int l = 0; l < NLVL; ++l) atomicAdd(&loss[l], com_s[l] * inv);
        atomicAdd(&loss[NLVL], recon_s * inv);
    }
}

// ---------------------------------------------------------------------------
extern "C" void kernel_launch(void* const* d_in, const int* in_sizes, int n_in,
                              void* d_out, int out_size, void* d_ws, size_t ws_size,
                              hipStream_t stream) {
    (void)in_sizes; (void)n_in; (void)out_size; (void)ws_size;
    const float* x   = (const float*)d_in[0];
    const float* Ws  = (const float*)d_in[1];
    const float* cbs = (const float*)d_in[2];
    float* out = (float*)d_out;

    // workspace layout (~273 KB): cbproj f32[4096] | cbn f32[128] | G bf16[131072]
    float*  cbproj = (float*)d_ws;
    float*  cbn    = cbproj + NLVL * CBSZ * CBSZ;
    __bf16* G      = (__bf16*)(cbn + NLVL * CBSZ);

    rvq_prep_cbproj<<<(NLVL * CBSZ * CBSZ * 32) / 256, 256, 0, stream>>>(Ws, cbs, cbproj);
    rvq_prep_G<<<(NLVL * CBSZ * DIM) / 256, 256, 0, stream>>>(Ws, cbproj, G);
    rvq_prep_cbn<<<1, 128, 0, stream>>>(cbproj, cbn);
    rvq_zero<<<1, 32, 0, stream>>>(out + (size_t)NBATCH * DIM + (size_t)NLVL * NBATCH);

    const size_t smem = (size_t)4 * 16 * DIM * sizeof(__bf16)
                      + (size_t)4 * 16 * 32 * sizeof(float)
                      + (size_t)4 * NLVL * 16 * sizeof(int);
    rvq_main<<<NBATCH / 64, 128, smem, stream>>>(x, cbs, G, cbn, out);
}
